// Transformer_4226247819620
// MI455X (gfx1250) — matrix operand order repaired and checked by reading, not compiled, not run on hardware
//
#include <hip/hip_runtime.h>
#include <math.h>

typedef __attribute__((ext_vector_type(16))) __bf16 v16bf;
typedef __attribute__((ext_vector_type(8)))  float  v8f;
typedef __attribute__((ext_vector_type(8)))  unsigned short ushort8;

__device__ __forceinline__ unsigned short f2bf(float f) {
  unsigned int u = __builtin_bit_cast(unsigned int, f);
  u += 0x7FFFu + ((u >> 16) & 1u);          // round-to-nearest-even
  return (unsigned short)(u >> 16);
}
__device__ __forceinline__ unsigned int pk2bf(float a, float b) {
  return (unsigned int)f2bf(a) | ((unsigned int)f2bf(b) << 16);
}
__device__ __forceinline__ float bf2f(unsigned short h) {
  unsigned int u = (unsigned int)h << 16;
  return __builtin_bit_cast(float, u);
}

union FragU { v16bf bf; ushort8 h[2]; };

// async copy of 32 contiguous bytes global -> LDS (CDNA5 async-LDS path).
// offset applies to BOTH the LDS and global address per ISA 08_async_tensor §4.4.
__device__ __forceinline__ void async_copy32(unsigned short* dst,
                                             const unsigned short* src)
{
  unsigned int       lds = (unsigned int)(unsigned long long)dst; // low 32b = LDS offset
  unsigned long long ga  = (unsigned long long)src;
  asm volatile(
      "global_load_async_to_lds_b128 %0, %1, off\n\t"
      "global_load_async_to_lds_b128 %0, %1, off offset:16"
      :: "v"(lds), "v"(ga) : "memory");
}
__device__ __forceinline__ void async_wait0()
{
  asm volatile("s_wait_asynccnt 0x0" ::: "memory");
}

// ---- bf16 staging: async fast path, scalar edge path ----
__device__ __forceinline__ void stageA_bf(unsigned short* __restrict__ As,
    const unsigned short* __restrict__ Ab, int rowBase, int k0, int lda,
    int M, int K, int tid)
{
  int r   = tid >> 1;
  int cst = (tid & 1) * 16;
  const unsigned short* src = Ab + (long long)(rowBase + r) * lda + (k0 + cst);
  unsigned short* dst = &As[r * 32 + cst];
  if (rowBase + 128 <= M && k0 + 32 <= K) {          // uniform
    async_copy32(dst, src);
  } else {
    bool rok = (rowBase + r) < M;
#pragma unroll
    for (int i = 0; i < 16; ++i)
      dst[i] = (rok && (k0 + cst + i) < K) ? src[i] : (unsigned short)0;
  }
}

__device__ __forceinline__ void stageB_nt_bf(unsigned short* __restrict__ Bs,
    const unsigned short* __restrict__ Bb, int colBase, int k0, int ldb,
    int N, int K, int tid)
{
  int n   = tid >> 1;
  int cst = (tid & 1) * 16;
  const unsigned short* src = Bb + (long long)(colBase + n) * ldb + (k0 + cst);
  unsigned short* dst = &Bs[n * 32 + cst];
  if (colBase + 128 <= N && k0 + 32 <= K) {          // uniform
    async_copy32(dst, src);
  } else {
    bool nok = (colBase + n) < N;
#pragma unroll
    for (int i = 0; i < 16; ++i)
      dst[i] = (nok && (k0 + cst + i) < K) ? src[i] : (unsigned short)0;
  }
}

__device__ __forceinline__ void stageB_nn_bf(unsigned short* __restrict__ Bs,
    const unsigned short* __restrict__ Bb, int colBase, int k0, int ldb,
    int N, int K, int tid)
{
  int kk  = tid >> 3;                   // 0..31
  int nst = (tid & 7) * 16;             // 0..112
  const unsigned short* src = Bb + (long long)(k0 + kk) * ldb + (colBase + nst);
  if (colBase + 128 <= N && k0 + 32 <= K) {          // uniform
    ushort8 a = *(const ushort8*)src;
    ushort8 b = *(const ushort8*)(src + 8);
#pragma unroll
    for (int i = 0; i < 8; ++i) Bs[(nst + i) * 32 + kk] = a[i];
#pragma unroll
    for (int i = 0; i < 8; ++i) Bs[(nst + 8 + i) * 32 + kk] = b[i];
  } else {
    bool kok = (k0 + kk) < K;
#pragma unroll
    for (int i = 0; i < 16; ++i) {
      int gn = colBase + nst + i;
      Bs[(nst + i) * 32 + kk] = (kok && gn < N) ? src[i] : (unsigned short)0;
    }
  }
}

// Generic batched bf16 GEMM: C = A[M,K] @ op(B) (+bias)(+residual)(gelu)
// flags: 1  = B stored [K][N] (NN; default is [N][K] NT)
//        2  = GELU epilogue
//        4  = add fp32 residual (read at same offset as C)
//        8  = causal block skip (QK^T scores)
//        16 = causal K clamp (att@V)
//        64 = store C as bf16 (else fp32)
__global__ __launch_bounds__(256) void gemm_wmma(
    const unsigned short* __restrict__ A, const unsigned short* __restrict__ B,
    const float* __restrict__ bias, const float* __restrict__ residual,
    void* __restrict__ C,
    int M, int N, int K, int lda, int ldb, int ldc,
    int batchDiv,
    long long aOuter, long long aInner,
    long long bOuter, long long bInner,
    long long cOuter, long long cInner,
    int flags)
{
  __shared__ unsigned short As[2][128 * 32];   // [m][k] bf16, double-buffered
  __shared__ unsigned short Bs[2][128 * 32];   // [n][k] bf16, double-buffered

  const int rowBase = blockIdx.y * 128;
  const int colBase = blockIdx.x * 128;
  if ((flags & 8) && colBase > rowBase + 127) return;   // fully-masked causal block

  int Keff = K;
  if (flags & 16) { int lim = rowBase + 128; Keff = lim < K ? lim : K; }

  const int tid      = threadIdx.x;
  const int lane     = tid & 31;
  const int wave     = tid >> 5;            // 8 waves
  const int waveM    = wave >> 2;           // 0..1 -> 64-row slab
  const int waveN    = wave & 3;            // 0..3 -> 32-col slab
  const int laneHalf = lane >> 4;
  const int lane16   = lane & 15;

  const long long z = blockIdx.z;
  const unsigned short* Ab = A + (z / batchDiv) * aOuter + (z % batchDiv) * aInner;
  const unsigned short* Bb = B + (z / batchDiv) * bOuter + (z % batchDiv) * bInner;
  const long long cOff = (z / batchDiv) * cOuter + (z % batchDiv) * cInner;
  float*          Cf = (float*)C + cOff;
  unsigned short* Ch = (unsigned short*)C + cOff;
  const float*    Rb = residual ? (residual + cOff) : nullptr;

  v8f acc[4][2] = {};

  stageA_bf(As[0], Ab, rowBase, 0, lda, M, Keff, tid);
  if (flags & 1) stageB_nn_bf(Bs[0], Bb, colBase, 0, ldb, N, Keff, tid);
  else           stageB_nt_bf(Bs[0], Bb, colBase, 0, ldb, N, Keff, tid);
  async_wait0();
  __syncthreads();

  int buf = 0;
  for (int k0 = 0; k0 < Keff; k0 += 32) {
    int nk = k0 + 32;
    if (nk < Keff) {
      stageA_bf(As[buf ^ 1], Ab, rowBase, nk, lda, M, Keff, tid);
      if (flags & 1) stageB_nn_bf(Bs[buf ^ 1], Bb, colBase, nk, ldb, N, Keff, tid);
      else           stageB_nt_bf(Bs[buf ^ 1], Bb, colBase, nk, ldb, N, Keff, tid);
    }

    const unsigned short* Ac = As[buf];
    const unsigned short* Bc = Bs[buf];
    FragU afrg[4], bfrg[2];
#pragma unroll
    for (int m = 0; m < 4; ++m) {
      int r = waveM * 64 + m * 16 + lane16;
      afrg[m].h[0] = *(const ushort8*)&Ac[r * 32 + laneHalf * 8];       // K 0..7 / 8..15
      afrg[m].h[1] = *(const ushort8*)&Ac[r * 32 + 16 + laneHalf * 8];  // K 16..23 / 24..31
    }
#pragma unroll
    for (int n = 0; n < 2; ++n) {
      int c = waveN * 32 + n * 16 + lane16;
      bfrg[n].h[0] = *(const ushort8*)&Bc[c * 32 + laneHalf * 8];
      bfrg[n].h[1] = *(const ushort8*)&Bc[c * 32 + 16 + laneHalf * 8];
    }
#pragma unroll
    for (int m = 0; m < 4; ++m)
#pragma unroll
      for (int n = 0; n < 2; ++n)
        acc[m][n] = __builtin_amdgcn_wmma_f32_16x16x32_bf16(
            false, afrg[m].bf, false, bfrg[n].bf, (short)0, acc[m][n],
            false, false);
    async_wait0();          // async DMA of next tile overlapped the WMMAs above
    __syncthreads();
    buf ^= 1;
  }

  // ---- epilogue: D layout = VGPR r -> M = r + 8*laneHalf, N = lane16 ----
  const bool interior = (rowBase + 128 <= M) && (colBase + 128 <= N);
  if (interior) {
#pragma unroll
    for (int m = 0; m < 4; ++m) {
      int rtile = rowBase + waveM * 64 + m * 16 + laneHalf * 8;
#pragma unroll
      for (int n = 0; n < 2; ++n) {
        int col = colBase + waveN * 32 + n * 16 + lane16;
        float bv = bias ? bias[col] : 0.0f;
#pragma unroll
        for (int r = 0; r < 8; ++r) {
          long long idx = (long long)(rtile + r) * ldc + col;
          float v = acc[m][n][r] + bv;
          if (flags & 4) v += Rb[idx];
          if (flags & 2) {
            float t = v + 0.044715f * v * v * v;
            v = 0.5f * v * (1.0f + tanhf(0.7978845608f * t));
          }
          if (flags & 64) Ch[idx] = f2bf(v);
          else            Cf[idx] = v;
        }
      }
    }
  } else {
#pragma unroll
    for (int m = 0; m < 4; ++m) {
      int rtile = rowBase + waveM * 64 + m * 16 + laneHalf * 8;
#pragma unroll
      for (int n = 0; n < 2; ++n) {
        int col = colBase + waveN * 32 + n * 16 + lane16;
        if (col < N) {
          float bv = bias ? bias[col] : 0.0f;
#pragma unroll
          for (int r = 0; r < 8; ++r) {
            int grow = rtile + r;
            if (grow < M) {
              long long idx = (long long)grow * ldc + col;
              float v = acc[m][n][r] + bv;
              if (flags & 4) v += Rb[idx];
              if (flags & 2) {
                float t = v + 0.044715f * v * v * v;
                v = 0.5f * v * (1.0f + tanhf(0.7978845608f * t));
              }
              if (flags & 64) Ch[idx] = f2bf(v);
              else            Cf[idx] = v;
            }
          }
        }
      }
    }
  }
}

// transpose-convert fp32 [K][N] -> bf16 [N][K]
__global__ __launch_bounds__(256) void cvt_t_k(
    const float* __restrict__ in, unsigned short* __restrict__ out, int K, int N)
{
  __shared__ float t[32][33];
  int nb = blockIdx.x * 32, kb = blockIdx.y * 32;
  int tx = threadIdx.x & 31, ty = threadIdx.x >> 5;
  for (int r = ty; r < 32; r += 8) {
    int k = kb + r, n = nb + tx;
    t[r][tx] = (k < K && n < N) ? in[(long long)k * N + n] : 0.f;
  }
  __syncthreads();
  for (int r = ty; r < 32; r += 8) {
    int n = nb + r, k = kb + tx;
    if (n < N && k < K) out[(long long)n * K + k] = f2bf(t[tx][r]);
  }
}

// elementwise fp32 -> bf16 (n must be multiple of 8 for the fast path)
__global__ __launch_bounds__(256) void cvt_bf_k(
    const float* __restrict__ in, unsigned short* __restrict__ out, long long n)
{
  long long i = ((long long)blockIdx.x * 256 + threadIdx.x) * 8;
  if (i + 8 <= n) {
    float4 f0 = *(const float4*)(in + i);
    float4 f1 = *(const float4*)(in + i + 4);
    unsigned int* o = (unsigned int*)(out + i);
    o[0] = pk2bf(f0.x, f0.y); o[1] = pk2bf(f0.z, f0.w);
    o[2] = pk2bf(f1.x, f1.y); o[3] = pk2bf(f1.z, f1.w);
  } else {
    for (long long j = i; j < n; ++j) out[j] = f2bf(in[j]);
  }
}

__global__ __launch_bounds__(256) void embed_k(
    const int* __restrict__ tok, const float* __restrict__ wte,
    const float* __restrict__ wpe, float* __restrict__ x, int Ss, int Dd)
{
  long long rs = blockIdx.x;
  int s = (int)(rs % Ss);
  long long t = tok[rs];
  for (int i = threadIdx.x; i < Dd; i += 256)
    x[rs * Dd + i] = wte[t * Dd + i] + wpe[(long long)s * Dd + i];
}

// LayerNorm fp32 in -> bf16 out
__global__ __launch_bounds__(256) void layernorm_bf_k(
    const float* __restrict__ x, const float* __restrict__ w,
    const float* __restrict__ b, unsigned short* __restrict__ out, int Dd)
{
  __shared__ float red[256];
  long long row = blockIdx.x;
  const float* xr = x + row * Dd;
  float s = 0.f;
  for (int i = threadIdx.x; i < Dd; i += 256) s += xr[i];
  red[threadIdx.x] = s; __syncthreads();
  for (int o = 128; o > 0; o >>= 1) {
    if (threadIdx.x < o) red[threadIdx.x] += red[threadIdx.x + o];
    __syncthreads();
  }
  float mean = red[0] / Dd;
  __syncthreads();
  float v = 0.f;
  for (int i = threadIdx.x; i < Dd; i += 256) { float d = xr[i] - mean; v += d * d; }
  red[threadIdx.x] = v; __syncthreads();
  for (int o = 128; o > 0; o >>= 1) {
    if (threadIdx.x < o) red[threadIdx.x] += red[threadIdx.x + o];
    __syncthreads();
  }
  float rstd = rsqrtf(red[0] / Dd + 1e-5f);
  unsigned short* orow = out + row * Dd;
  for (int i = threadIdx.x; i < Dd; i += 256)
    orow[i] = f2bf((xr[i] - mean) * rstd * w[i] + b[i]);
}

// causal scaled softmax over bf16 scores; zeroes only up to next 128 boundary
// (the att@V GEMM K-clamp never reads past it)
__global__ __launch_bounds__(256) void softmax_causal_bf_k(
    unsigned short* __restrict__ scores, int Ss, float scale)
{
  __shared__ float red[256];
  long long idx = blockIdx.x;                // (b*H + h)*S + q
  int q = (int)(idx % Ss);
  unsigned short* row = scores + idx * Ss;
  int len = q + 1;
  float m = -1e30f;
  for (int j = threadIdx.x; j < len; j += 256) m = fmaxf(m, bf2f(row[j]) * scale);
  red[threadIdx.x] = m; __syncthreads();
  for (int o = 128; o > 0; o >>= 1) {
    if (threadIdx.x < o) red[threadIdx.x] = fmaxf(red[threadIdx.x], red[threadIdx.x + o]);
    __syncthreads();
  }
  float rmax = red[0];
  __syncthreads();
  float s = 0.f;
  for (int j = threadIdx.x; j < len; j += 256) {
    float e = __expf(bf2f(row[j]) * scale - rmax);
    row[j] = f2bf(e);
    s += e;
  }
  red[threadIdx.x] = s; __syncthreads();
  for (int o = 128; o > 0; o >>= 1) {
    if (threadIdx.x < o) red[threadIdx.x] += red[threadIdx.x + o];
    __syncthreads();
  }
  float inv = 1.0f / red[0];
  for (int j = threadIdx.x; j < len; j += 256) row[j] = f2bf(bf2f(row[j]) * inv);
  int lim = (q & ~127) + 128; if (lim > Ss) lim = Ss;
  for (int j = len + threadIdx.x; j < lim; j += 256) row[j] = 0;
}

extern "C" void kernel_launch(void* const* d_in, const int* in_sizes, int n_in,
                              void* d_out, int out_size, void* d_ws, size_t ws_size,
                              hipStream_t stream)
{
  (void)in_sizes; (void)n_in; (void)out_size; (void)ws_size;
  constexpr int L = 4, Hh = 12, Dd = 768, Vv = 50257, Ss = 1024, Bb = 2, FFf = 3072;
  constexpr int BS = Bb * Ss;

  const int*   tok    = (const int*)  d_in[0];
  const float* wte    = (const float*)d_in[1];
  const float* wpe    = (const float*)d_in[2];
  const float* ln1w   = (const float*)d_in[3];
  const float* ln1b   = (const float*)d_in[4];
  const float* attnw  = (const float*)d_in[5];
  const float* attnb  = (const float*)d_in[6];
  const float* aprojw = (const float*)d_in[7];
  const float* aprojb = (const float*)d_in[8];
  const float* ln2w   = (const float*)d_in[9];
  const float* ln2b   = (const float*)d_in[10];
  const float* fcw    = (const float*)d_in[11];
  const float* fcb    = (const float*)d_in[12];
  const float* mprojw = (const float*)d_in[13];
  const float* mprojb = (const float*)d_in[14];
  const float* lnfw   = (const float*)d_in[15];
  const float* lnfb   = (const float*)d_in[16];
  const float* outw   = (const float*)d_in[17];
  float* logits = (float*)d_out;

  // workspace layout
  float* x = (float*)d_ws;                                   // fp32 residual stream
  unsigned short* u = (unsigned short*)(x + (size_t)BS * Dd);
  unsigned short* h_bf   = u; u += (size_t)BS * Dd;
  unsigned short* qkv_bf = u; u += (size_t)BS * 3 * Dd;
  unsigned short* sc_bf  = u; u += (size_t)Bb * Hh * Ss * Ss;
  unsigned short* y_bf   = u; u += (size_t)BS * Dd;
  unsigned short* ff_bf  = u; u += (size_t)BS * FFf;
  unsigned short* wq_bf  = u; u += (size_t)L * 3 * Dd * Dd;  // [3D][D]
  unsigned short* wa_bf  = u; u += (size_t)L * Dd * Dd;      // [D][D]
  unsigned short* wf_bf  = u; u += (size_t)L * FFf * Dd;     // [FF][D]
  unsigned short* wm_bf  = u; u += (size_t)L * Dd * FFf;     // [D][FF]
  unsigned short* wo_bf  = u;                                // [V][D]

  dim3 blk(256);
  const float scale = 0.125f;  // 1/sqrt(HD=64)

  // ---- per-launch weight convert + transpose to bf16 [N][K] ----
  for (int l = 0; l < L; ++l) {
    cvt_t_k<<<dim3(3 * Dd / 32, Dd / 32), blk, 0, stream>>>(
        attnw + (size_t)l * Dd * 3 * Dd, wq_bf + (size_t)l * 3 * Dd * Dd, Dd, 3 * Dd);
    cvt_t_k<<<dim3(Dd / 32, Dd / 32), blk, 0, stream>>>(
        aprojw + (size_t)l * Dd * Dd, wa_bf + (size_t)l * Dd * Dd, Dd, Dd);
    cvt_t_k<<<dim3(FFf / 32, Dd / 32), blk, 0, stream>>>(
        fcw + (size_t)l * Dd * FFf, wf_bf + (size_t)l * FFf * Dd, Dd, FFf);
    cvt_t_k<<<dim3(Dd / 32, FFf / 32), blk, 0, stream>>>(
        mprojw + (size_t)l * FFf * Dd, wm_bf + (size_t)l * Dd * FFf, FFf, Dd);
  }
  {
    long long n = (long long)Vv * Dd;
    cvt_bf_k<<<(unsigned)((n / 8 + 255) / 256), blk, 0, stream>>>(outw, wo_bf, n);
  }

  embed_k<<<BS, blk, 0, stream>>>(tok, wte, wpe, x, Ss, Dd);

  for (int l = 0; l < L; ++l) {
    layernorm_bf_k<<<BS, blk, 0, stream>>>(x, ln1w + l * Dd, ln1b + l * Dd, h_bf, Dd);

    // qkv = h @ attn_w + attn_b   (2048 x 2304, K=768) -> bf16
    gemm_wmma<<<dim3(3 * Dd / 128, BS / 128, 1), blk, 0, stream>>>(
        h_bf, wq_bf + (size_t)l * 3 * Dd * Dd, attnb + (size_t)l * 3 * Dd, nullptr,
        qkv_bf, BS, 3 * Dd, Dd, Dd, Dd, 3 * Dd,
        1, 0, 0, 0, 0, 0, 0, 64);

    // scores[b,h] = q @ k^T (1024x1024, K=64), causal blocks skipped -> bf16
    gemm_wmma<<<dim3(Ss / 128, Ss / 128, Bb * Hh), blk, 0, stream>>>(
        qkv_bf, qkv_bf + Dd, nullptr, nullptr, sc_bf,
        Ss, Ss, 64, 3 * Dd, 3 * Dd, Ss,
        Hh, (long long)Ss * 3 * Dd, 64, (long long)Ss * 3 * Dd, 64,
        (long long)Hh * Ss * Ss, (long long)Ss * Ss,
        8 | 64);

    softmax_causal_bf_k<<<Bb * Hh * Ss, blk, 0, stream>>>(sc_bf, Ss, scale);

    // y[b,h] = att @ v (1024 x 64, K causally clamped) -> bf16
    gemm_wmma<<<dim3(1, Ss / 128, Bb * Hh), blk, 0, stream>>>(
        sc_bf, qkv_bf + 2 * Dd, nullptr, nullptr, y_bf,
        Ss, 64, Ss, Ss, 3 * Dd, Dd,
        Hh, (long long)Hh * Ss * Ss, (long long)Ss * Ss,
        (long long)Ss * 3 * Dd, 64,
        (long long)Ss * Dd, 64,
        1 | 16 | 64);

    // x = x + y @ aproj_w + aproj_b  (fp32 out)
    gemm_wmma<<<dim3(Dd / 128, BS / 128, 1), blk, 0, stream>>>(
        y_bf, wa_bf + (size_t)l * Dd * Dd, aprojb + (size_t)l * Dd, x, x,
        BS, Dd, Dd, Dd, Dd, Dd,
        1, 0, 0, 0, 0, 0, 0, 4);

    layernorm_bf_k<<<BS, blk, 0, stream>>>(x, ln2w + l * Dd, ln2b + l * Dd, h_bf, Dd);

    // ff = gelu(h @ fc_w + fc_b)   (2048 x 3072) -> bf16
    gemm_wmma<<<dim3(FFf / 128, BS / 128, 1), blk, 0, stream>>>(
        h_bf, wf_bf + (size_t)l * FFf * Dd, fcb + (size_t)l * FFf, nullptr, ff_bf,
        BS, FFf, Dd, Dd, Dd, FFf,
        1, 0, 0, 0, 0, 0, 0, 2 | 64);

    // x = x + ff @ mproj_w + mproj_b   (K=3072, fp32 out)
    gemm_wmma<<<dim3(Dd / 128, BS / 128, 1), blk, 0, stream>>>(
        ff_bf, wm_bf + (size_t)l * Dd * FFf, mprojb + (size_t)l * Dd, x, x,
        BS, Dd, FFf, FFf, FFf, Dd,
        1, 0, 0, 0, 0, 0, 0, 4);
  }

  layernorm_bf_k<<<BS, blk, 0, stream>>>(x, lnfw, lnfb, h_bf, Dd);

  // logits = h @ out_w^T   (2048 x 50257, K=768, fp32 out)
  gemm_wmma<<<dim3((Vv + 127) / 128, BS / 128, 1), blk, 0, stream>>>(
      h_bf, wo_bf, nullptr, nullptr, logits,
      BS, Vv, Dd, Dd, Dd, Vv,
      1, 0, 0, 0, 0, 0, 0, 0);
}
